// multi_head_MLP_35287451304289
// MI455X (gfx1250) — matrix operand order repaired and checked by reading, not compiled, not run on hardware
//
#include <hip/hip_runtime.h>
#include <hip/hip_bf16.h>

typedef __attribute__((ext_vector_type(16))) _Float16 v16h;
typedef __attribute__((ext_vector_type(8)))  _Float16 v8h;
typedef __attribute__((ext_vector_type(8)))  float    v8f;

// ---------------- packed-weight geometry (units: f16 halves) ----------------
// One 32x16 (KxN) B tile = 512 halves; lane l holds col n=l%16, K-run (l/16)*16+e.
static constexpr long SZ_W1 = 1L  * 32 * 512;  // K30->32, N512
static constexpr long SZ_W2 = 16L * 32 * 512;  // 512x512
static constexpr long SZ_W3 = 16L * 31 * 512;  // 512x482->496
static constexpr long SZ_W4 = 16L * 32 * 512;  // 512x512
static constexpr long SZ_H1 = 16L * 8  * 512;  // 512x128 (per head)
static constexpr long SZ_H2 = 4L  * 4  * 512;  // 128x64
static constexpr long SZ_H3 = 2L  * 2  * 512;  // 64x32
static constexpr long SZ_H4 = 1L  * 1  * 512;  // 32x16
static constexpr long SZ_H5 = 1L  * 1  * 512;  // 16->32 x 1->16 (zero padded)
static constexpr long OFF_W1 = 0;
static constexpr long OFF_W2 = OFF_W1 + SZ_W1;
static constexpr long OFF_W3 = OFF_W2 + SZ_W2;
static constexpr long OFF_W4 = OFF_W3 + SZ_W3;
static constexpr long OFF_H1 = OFF_W4 + SZ_W4;
static constexpr long OFF_H2 = OFF_H1 + 8 * SZ_H1;
static constexpr long OFF_H3 = OFF_H2 + 8 * SZ_H2;
static constexpr long OFF_H4 = OFF_H3 + 8 * SZ_H3;
static constexpr long OFF_H5 = OFF_H4 + 8 * SZ_H4;

// ---------------- weight repack: fp32 [G,K,N] -> f16 WMMA-B tiles ----------
// 3D launch: blockIdx = (kt, nt, g), 512 threads = one 32x16 tile. No div/mod.
__global__ void pack_weights(const float* __restrict__ W,
                             _Float16* __restrict__ dst, int K, int N) {
  const int kt = blockIdx.x, nt = blockIdx.y, g = blockIdx.z;
  const int KT = gridDim.x, NT = gridDim.y;
  const int w    = threadIdx.x;          // 0..511
  const int lane = w >> 4;
  const int e    = w & 15;
  const int n = nt * 16 + (lane & 15);
  const int k = kt * 32 + (lane >> 4) * 16 + e;  // B layout: lanes 16-31 hold K+16
  float v = 0.f;
  if (k < K && n < N) v = W[(long)g * K * N + (long)k * N + n];
  dst[((long)g * NT * KT + (long)nt * KT + kt) * 512 + w] = (_Float16)v;
}

// ---------------- WMMA helpers ----------------
__device__ __forceinline__ v8f wmma_f16(v16h a, v16h b, v8f c) {
  return __builtin_amdgcn_wmma_f32_16x16x32_f16(false, a, false, b, (short)0, c,
                                                false, false);
}

// A-operand loader: lane holds row m=l%16; 16 halves = K+(l/16)*16+e
// (contiguous, matching packed B).
template <int KT>
__device__ __forceinline__ void load_afrags(v16h* afr, const _Float16* buf,
                                            int strideH, int lane) {
  const _Float16* rb = buf + (lane & 15) * strideH + (lane >> 4) * 16;
#pragma unroll
  for (int kt = 0; kt < KT; ++kt) {
    v8h lo = *(const v8h*)(rb + kt * 32);
    v8h hi = *(const v8h*)(rb + kt * 32 + 8);
    afr[kt] = __builtin_shufflevector(lo, hi, 0, 1, 2, 3, 4, 5, 6, 7, 8, 9, 10,
                                      11, 12, 13, 14, 15);
  }
}

template <int KT>
__device__ __forceinline__ v8f gemm_tile(const v16h* afr,
                                         const _Float16* __restrict__ wt,
                                         int nt, float bias, int lane) {
  v8f acc = {bias, bias, bias, bias, bias, bias, bias, bias};
  const _Float16* bp = wt + (long)nt * KT * 512 + lane * 16;
#pragma unroll
  for (int kt = 0; kt < KT; ++kt) {
    v16h b = *(const v16h*)(bp + (long)kt * 512);   // 32B contiguous per lane
    acc = wmma_f16(afr[kt], b, acc);
  }
  return acc;
}

// D layout: lane holds col n=l%16, rows m=(l/16)*8+r  -> scatter to row-major LDS
__device__ __forceinline__ void store_d(_Float16* buf, int strideH, int nt,
                                        v8f acc, bool relu, int nmax, int lane) {
  int col = nt * 16 + (lane & 15);
  if (col >= nmax) return;
  int m0 = (lane >> 4) * 8;
#pragma unroll
  for (int r = 0; r < 8; ++r) {
    float v = acc[r];
    if (relu) v = (v > 0.f) ? v : 0.f;   // single v_max_num_f32
    buf[(m0 + r) * strideH + col] = (_Float16)v;
  }
}

template <int KT>
__device__ __forceinline__ void run_layer(const v16h* afr,
                                          const _Float16* __restrict__ wt,
                                          const float* __restrict__ bias,
                                          _Float16* dst, int strideH, int NT,
                                          int nmax, bool relu, int lane) {
  for (int nt = 0; nt < NT; ++nt) {
    int n = nt * 16 + (lane & 15);
    float bv = (n < nmax) ? bias[n] : 0.f;
    v8f acc = gemm_tile<KT>(afr, wt, nt, bv, lane);
    store_d(dst, strideH, nt, acc, relu, nmax, lane);
  }
}

// ---------------- fused multi-head MLP ----------------
// 2 waves / block, each wave owns a 16-row tile. LDS ~46KB static.
__global__ void __launch_bounds__(64) mlp_fused(
    const float* __restrict__ x,
    const float* __restrict__ b1, const float* __restrict__ b2,
    const float* __restrict__ b3, const float* __restrict__ b4,
    const float* __restrict__ hb1, const float* __restrict__ hb2,
    const float* __restrict__ hb3, const float* __restrict__ hb4,
    const float* __restrict__ hb5,
    const _Float16* __restrict__ wp, float* __restrict__ out) {
  __shared__ __align__(16) _Float16 s_act[2][16][512];
  __shared__ __align__(16) _Float16 s_xsk[2][16][32];
  __shared__ __align__(16) _Float16 s_p[2][16][128];
  __shared__ __align__(16) _Float16 s_q[2][16][64];

  const int lane = threadIdx.x & 31;
  const int wv   = threadIdx.x >> 5;
  const int row0 = blockIdx.x * 32 + wv * 16;

  _Float16(*act)[512] = s_act[wv];
  _Float16(*xsk)[32]  = s_xsk[wv];
  _Float16(*hp)[128]  = s_p[wv];
  _Float16(*hq)[64]   = s_q[wv];

  // ---- build x_skip = [x, sin x, cos x], K padded 30->32 with zeros ----
  {
    int cc = lane;  // each lane owns one column, walks 16 rows
    for (int rr = 0; rr < 16; ++rr) {
      _Float16 v = (_Float16)0.f;
      if (cc < 30) {
        float xv = x[(long)(row0 + rr) * 10 + (cc % 10)];
        int grp = cc / 10;
        float f = (grp == 0) ? xv : (grp == 1 ? __sinf(xv) : __cosf(xv));
        v = (_Float16)f;
      }
      xsk[rr][cc] = v;
    }
  }
  __syncthreads();

  // ---- L1: 30(32) -> 512, relu ----
  v16h a1[1];
  load_afrags<1>(a1, &xsk[0][0], 32, lane);
  run_layer<1>(a1, wp + OFF_W1, b1, &act[0][0], 512, 32, 512, true, lane);
  __syncthreads();

  v16h af[16];
  // ---- L2: 512 -> 512, relu ----
  load_afrags<16>(af, &act[0][0], 512, lane);
  run_layer<16>(af, wp + OFF_W2, b2, &act[0][0], 512, 32, 512, true, lane);
  __syncthreads();

  // ---- L3: 512 -> 482 (relu fused: relu hits whole concat next) ----
  load_afrags<16>(af, &act[0][0], 512, lane);
  run_layer<16>(af, wp + OFF_W3, b3, &act[0][0], 512, 31, 482, true, lane);
  // concat: cols 482..511 = relu(x_skip)
  for (int i = lane; i < 16 * 30; i += 32) {
    int rr = i / 30, cc = i % 30;
    float v = (float)xsk[rr][cc];
    act[rr][482 + cc] = (_Float16)((v > 0.f) ? v : 0.f);
  }
  __syncthreads();

  // ---- L4: 512 -> 512, NO relu (reference discards it) ----
  load_afrags<16>(af, &act[0][0], 512, lane);
  run_layer<16>(af, wp + OFF_W4, b4, &act[0][0], 512, 32, 512, false, lane);
  __syncthreads();

  // ---- h kept entirely in registers for all 8 heads (128 VGPRs) ----
  load_afrags<16>(af, &act[0][0], 512, lane);

  for (int g = 0; g < 8; ++g) {
    // H1: 512 -> 128, relu
    run_layer<16>(af, wp + OFF_H1 + (long)g * SZ_H1, hb1 + g * 128, &hp[0][0],
                  128, 8, 128, true, lane);
    __syncthreads();
    v16h y1[4];
    load_afrags<4>(y1, &hp[0][0], 128, lane);
    // H2: 128 -> 64, relu
    run_layer<4>(y1, wp + OFF_H2 + (long)g * SZ_H2, hb2 + g * 64, &hq[0][0], 64,
                 4, 64, true, lane);
    __syncthreads();
    v16h y2[2];
    load_afrags<2>(y2, &hq[0][0], 64, lane);
    // H3: 64 -> 32, relu (into cols 0..31 of hp)
    run_layer<2>(y2, wp + OFF_H3 + (long)g * SZ_H3, hb3 + g * 32, &hp[0][0],
                 128, 2, 32, true, lane);
    __syncthreads();
    v16h y3[1];
    load_afrags<1>(y3, &hp[0][0], 128, lane);
    // H4: 32 -> 16, relu (into cols 0..15 of hq; cols 16..31 stale*0-weights)
    run_layer<1>(y3, wp + OFF_H4 + (long)g * SZ_H4, hb4 + g * 16, &hq[0][0], 64,
                 1, 16, true, lane);
    __syncthreads();
    v16h y4[1];
    load_afrags<1>(y4, &hq[0][0], 64, lane);
    // H5: 16(pad32) -> 1 (col 0 of zero-padded 16-wide tile), no relu
    v8f acc = {0.f, 0.f, 0.f, 0.f, 0.f, 0.f, 0.f, 0.f};
    {
      v16h b = *(const v16h*)(wp + OFF_H5 + (long)g * SZ_H5 + lane * 16);
      acc = wmma_f16(y4[0], b, acc);
    }
    if ((lane & 15) == 0) {
      int m0 = (lane >> 4) * 8;
      float bb = hb5[g];
#pragma unroll
      for (int r = 0; r < 8; ++r)
        out[(long)(row0 + m0 + r) * 8 + g] = acc[r] + bb;
    }
    __syncthreads();
  }
}

// ---------------- launch ----------------
extern "C" void kernel_launch(void* const* d_in, const int* in_sizes, int n_in,
                              void* d_out, int out_size, void* d_ws,
                              size_t ws_size, hipStream_t stream) {
  (void)in_sizes; (void)n_in; (void)out_size; (void)ws_size;
  const float* x   = (const float*)d_in[0];
  const float* W1  = (const float*)d_in[1];
  const float* b1  = (const float*)d_in[2];
  const float* W2  = (const float*)d_in[3];
  const float* b2  = (const float*)d_in[4];
  const float* W3  = (const float*)d_in[5];
  const float* b3  = (const float*)d_in[6];
  const float* W4  = (const float*)d_in[7];
  const float* b4  = (const float*)d_in[8];
  const float* HW1 = (const float*)d_in[9];
  const float* Hb1 = (const float*)d_in[10];
  const float* HW2 = (const float*)d_in[11];
  const float* Hb2 = (const float*)d_in[12];
  const float* HW3 = (const float*)d_in[13];
  const float* Hb3 = (const float*)d_in[14];
  const float* HW4 = (const float*)d_in[15];
  const float* Hb4 = (const float*)d_in[16];
  const float* HW5 = (const float*)d_in[17];
  const float* Hb5 = (const float*)d_in[18];

  _Float16* wp = (_Float16*)d_ws;

  auto pack = [&](const float* W, long off, int K, int N, int KT, int NT, int G) {
    pack_weights<<<dim3(KT, NT, G), 512, 0, stream>>>(W, wp + off, K, N);
  };
  pack(W1,  OFF_W1,  30, 512,  1, 32, 1);
  pack(W2,  OFF_W2, 512, 512, 16, 32, 1);
  pack(W3,  OFF_W3, 512, 482, 16, 31, 1);
  pack(W4,  OFF_W4, 512, 512, 16, 32, 1);
  pack(HW1, OFF_H1, 512, 128, 16,  8, 8);
  pack(HW2, OFF_H2, 128,  64,  4,  4, 8);
  pack(HW3, OFF_H3,  64,  32,  2,  2, 8);
  pack(HW4, OFF_H4,  32,  16,  1,  1, 8);
  pack(HW5, OFF_H5,  16,   1,  1,  1, 8);

  const int B = 131072;
  mlp_fused<<<B / 32, 64, 0, stream>>>(x, b1, b2, b3, b4, Hb1, Hb2, Hb3, Hb4,
                                       Hb5, wp, (float*)d_out);
}
